// ObjectAttentionBlock_62423054680122
// MI455X (gfx1250) — matrix operand order repaired and checked by reading, not compiled, not run on hardware
//
#include <hip/hip_runtime.h>
#include <math.h>

// ---------------------------------------------------------------------------
// Types for CDNA5 WMMA (wave32): v_wmma_f32_16x16x32_bf16
// ---------------------------------------------------------------------------
typedef __bf16  v16bf __attribute__((ext_vector_type(16)));
typedef float   v8f   __attribute__((ext_vector_type(8)));

#define C_CH   512
#define K_REG  256
#define NB     8
#define HW     16384

// ---------------------------------------------------------------------------
// fp32 -> bf16 conversion (grid-stride)
// ---------------------------------------------------------------------------
__global__ void cvt_f32_bf16(const float* __restrict__ s, __bf16* __restrict__ d,
                             long long n) {
    long long i = (long long)blockIdx.x * blockDim.x + threadIdx.x;
    long long stride = (long long)gridDim.x * blockDim.x;
    for (; i < n; i += stride) d[i] = (__bf16)s[i];
}

// ---------------------------------------------------------------------------
// Batched bf16 GEMM + fused epilogue.
//   HAS_SCALE: D[m][n] = relu( scale[m] * sum_k A[m][k]*B[k][n] + bias[m] )
//   else:      D[m][n] = alpha * sum_k A[m][k]*B[k][n]
// A: [M][K] row-major (K contiguous), B: [K][N] row-major (N contiguous).
// TRANS==0: out[m*N+n]; TRANS==1: out[n*M+m] (vectorized b128 stores).
// Block tile 128x128, BK=64, 256 threads (8 waves), wave tile 32x64 (2x4 WMMA).
// Double-buffered LDS (one barrier per K-stage); B is software-pipelined
// through 16 VGPRs (HBM stream), A (L2-resident weights) copied directly.
// Requires M%128==0, N%128==0, K%64==0 (true for all uses here).
// ---------------------------------------------------------------------------
template<int TRANS, int HAS_SCALE, typename OutT>
__global__ __launch_bounds__(256, 1)
void gemm_cba(const __bf16* __restrict__ A, const __bf16* __restrict__ B,
              OutT* __restrict__ out, int M, int N, int K,
              long long sA, long long sB, long long sO,
              const float* __restrict__ scale, const float* __restrict__ bias,
              float alpha) {
    const int tid  = threadIdx.x;
    const int wid  = tid >> 5;
    const int lane = tid & 31;
    const int l16  = lane & 15;
    const int lh   = lane >> 4;          // half-wave select
    const int wm   = (wid & 3) * 32;     // wave M offset within tile
    const int wn   = (wid >> 2) * 64;    // wave N offset within tile
    const int m0   = blockIdx.x * 128;
    const int n0   = blockIdx.y * 128;

    A   += (long long)blockIdx.z * sA;
    B   += (long long)blockIdx.z * sB;
    out += (long long)blockIdx.z * sO;

    __shared__ __align__(16) __bf16 As[2][128][72];   // [m][k], 144B rows
    __shared__ __align__(16) __bf16 Bs[2][128][72];   // [n][k] (transposed tile)

    v8f acc[2][4];
#pragma unroll
    for (int mt = 0; mt < 2; ++mt)
#pragma unroll
        for (int nt = 0; nt < 4; ++nt)
            acc[mt][nt] = (v8f){0.f,0.f,0.f,0.f,0.f,0.f,0.f,0.f};

    const int arow = tid >> 1;           // 0..127
    const int acol = (tid & 1) * 32;     // 0 or 32
    const int bkr  = tid >> 4;           // 0..15
    const int bnc  = (tid & 15) * 8;     // 0..120

    // ---- prologue: fill buffer 0 ----
    {
        const __bf16* ag = A + (long long)(m0 + arow) * K + acol;
#pragma unroll
        for (int i = 0; i < 4; ++i)
            *reinterpret_cast<uint4*>(&As[0][arow][acol + i * 8]) =
                *reinterpret_cast<const uint4*>(ag + i * 8);
#pragma unroll
        for (int p = 0; p < 4; ++p) {
            uint4 t = *reinterpret_cast<const uint4*>(
                B + (long long)(bkr + p * 16) * N + (n0 + bnc));
            __bf16 tv[8];
            *reinterpret_cast<uint4*>(tv) = t;
#pragma unroll
            for (int e = 0; e < 8; ++e) Bs[0][bnc + e][bkr + p * 16] = tv[e];
        }
    }
    __syncthreads();

    for (int k0 = 0; k0 < K; k0 += 64) {
        const int  buf  = (k0 >> 6) & 1;
        const int  kn   = k0 + 64;
        const bool more = kn < K;

        uint4 rb[4];
        if (more) {                       // prefetch next B stage (HBM stream)
#pragma unroll
            for (int p = 0; p < 4; ++p)
                rb[p] = *reinterpret_cast<const uint4*>(
                    B + (long long)(kn + bkr + p * 16) * N + (n0 + bnc));
        }

        // ---- 2 k-chunks of 32, 8 WMMAs each (overlaps B prefetch latency) ----
#pragma unroll
        for (int kk = 0; kk < 64; kk += 32) {
            v16bf af[2], bfr[4];
#pragma unroll
            for (int mt = 0; mt < 2; ++mt) {
                const int r = wm + mt * 16 + l16;
                union { v16bf v; uint4 q[2]; } u;
                u.q[0] = *reinterpret_cast<const uint4*>(&As[buf][r][kk + lh * 8]);
                u.q[1] = *reinterpret_cast<const uint4*>(&As[buf][r][kk + 16 + lh * 8]);
                af[mt] = u.v;
            }
#pragma unroll
            for (int nt = 0; nt < 4; ++nt) {
                const int r = wn + nt * 16 + l16;
                union { v16bf v; uint4 q[2]; } u;
                u.q[0] = *reinterpret_cast<const uint4*>(&Bs[buf][r][kk + lh * 8]);
                u.q[1] = *reinterpret_cast<const uint4*>(&Bs[buf][r][kk + 16 + lh * 8]);
                bfr[nt] = u.v;
            }
#pragma unroll
            for (int mt = 0; mt < 2; ++mt)
#pragma unroll
                for (int nt = 0; nt < 4; ++nt)
                    acc[mt][nt] = __builtin_amdgcn_wmma_f32_16x16x32_bf16(
                        false, af[mt], false, bfr[nt],
                        (short)0, acc[mt][nt], false, false);
        }

        if (more) {                       // commit next stage into other buffer
            // A: direct global->LDS (L2-resident weight tile, short-lived regs)
            const __bf16* ag = A + (long long)(m0 + arow) * K + kn + acol;
#pragma unroll
            for (int i = 0; i < 4; ++i)
                *reinterpret_cast<uint4*>(&As[buf ^ 1][arow][acol + i * 8]) =
                    *reinterpret_cast<const uint4*>(ag + i * 8);
            // B: commit prefetched registers with transpose scatter
#pragma unroll
            for (int p = 0; p < 4; ++p) {
                __bf16 tv[8];
                *reinterpret_cast<uint4*>(tv) = rb[p];
#pragma unroll
                for (int e = 0; e < 8; ++e) Bs[buf ^ 1][bnc + e][bkr + p * 16] = tv[e];
            }
        }
        __syncthreads();                  // single barrier per K-stage
    }

    // ---- epilogue: BN-fold + ReLU (or alpha), vectorized transposed stores ----
#pragma unroll
    for (int mt = 0; mt < 2; ++mt) {
        const int mbase = m0 + wm + mt * 16 + lh * 8;   // 8 consecutive m per lane
        float sc[8], bi[8];
        if (HAS_SCALE) {
            *reinterpret_cast<float4*>(&sc[0]) = *reinterpret_cast<const float4*>(&scale[mbase]);
            *reinterpret_cast<float4*>(&sc[4]) = *reinterpret_cast<const float4*>(&scale[mbase + 4]);
            *reinterpret_cast<float4*>(&bi[0]) = *reinterpret_cast<const float4*>(&bias[mbase]);
            *reinterpret_cast<float4*>(&bi[4]) = *reinterpret_cast<const float4*>(&bias[mbase + 4]);
        }
#pragma unroll
        for (int nt = 0; nt < 4; ++nt) {
            const int n = n0 + wn + nt * 16 + l16;
            float v[8];
#pragma unroll
            for (int i = 0; i < 8; ++i) {
                float t = acc[mt][nt][i];
                if (HAS_SCALE) t = fmaxf(fmaf(sc[i], t, bi[i]), 0.f);
                else           t *= alpha;
                v[i] = t;
            }
            if (TRANS) {
                // 8 consecutive elements at out[n*M + mbase] -> b128 store(s)
                union { OutT t[8]; uint4 q[(sizeof(OutT) * 8) / 16]; } st;
#pragma unroll
                for (int i = 0; i < 8; ++i) st.t[i] = (OutT)v[i];
                uint4* dst = reinterpret_cast<uint4*>(&out[(long long)n * M + mbase]);
#pragma unroll
                for (int c = 0; c < (int)((sizeof(OutT) * 8) / 16); ++c) dst[c] = st.q[c];
            } else {
#pragma unroll
                for (int i = 0; i < 8; ++i)
                    out[(long long)(mbase + i) * N + n] = (OutT)v[i];
            }
        }
    }
}

// ---------------------------------------------------------------------------
// Row softmax over 256 columns, one wave32 per row (8 elems/lane).
// ---------------------------------------------------------------------------
__global__ __launch_bounds__(256)
void softmax256(const __bf16* __restrict__ in, __bf16* __restrict__ outp) {
    const int wid  = threadIdx.x >> 5;
    const int lane = threadIdx.x & 31;
    const long long row = (long long)blockIdx.x * 8 + wid;
    const long long off = row * 256 + lane * 8;

    __bf16 v[8];
    *reinterpret_cast<uint4*>(v) = *reinterpret_cast<const uint4*>(in + off);

    float f[8], mx = -INFINITY;
#pragma unroll
    for (int i = 0; i < 8; ++i) { f[i] = (float)v[i]; mx = fmaxf(mx, f[i]); }
#pragma unroll
    for (int o = 16; o > 0; o >>= 1) mx = fmaxf(mx, __shfl_xor(mx, o, 32));

    float sum = 0.f;
#pragma unroll
    for (int i = 0; i < 8; ++i) { f[i] = __expf(f[i] - mx); sum += f[i]; }
#pragma unroll
    for (int o = 16; o > 0; o >>= 1) sum += __shfl_xor(sum, o, 32);

    const float inv = 1.0f / sum;
#pragma unroll
    for (int i = 0; i < 8; ++i) v[i] = (__bf16)(f[i] * inv);
    *reinterpret_cast<uint4*>(outp + off) = *reinterpret_cast<const uint4*>(v);
}

// ---------------------------------------------------------------------------
// Host-side orchestration
// ---------------------------------------------------------------------------
extern "C" void kernel_launch(void* const* d_in, const int* in_sizes, int n_in,
                              void* d_out, int out_size, void* d_ws, size_t ws_size,
                              hipStream_t stream) {
    const float* x     = (const float*)d_in[0];  // [8,512,128,128]
    const float* proxy = (const float*)d_in[1];  // [8,512,256,1]
    const float* W     = (const float*)d_in[2];  // [7,512,512]
    const float* s     = (const float*)d_in[3];  // [7,512]
    const float* b     = (const float*)d_in[4];  // [7,512]
    float* outp        = (float*)d_out;          // [8,512,128,128]

    char* ws = (char*)d_ws;
    size_t off = 0;
    auto take = [&](size_t bytes) -> char* {
        char* p = ws + off;
        off += (bytes + 255) & ~(size_t)255;
        return p;
    };

    const long long nX   = (long long)NB * C_CH * HW;      // 67,108,864
    const long long nW   = 7LL * C_CH * C_CH;              // 1,835,008
    const long long nP   = (long long)NB * C_CH * K_REG;   // 1,048,576
    const long long nSim = (long long)NB * HW * K_REG;     // 33,554,432

    __bf16* Wb     = (__bf16*)take(nW * 2);
    __bf16* bufX   = (__bf16*)take(nX * 2);      // xb -> qT -> ctxT
    __bf16* bufT   = (__bf16*)take(nX * 2);      // tmp1 -> sim(lo)/simsm(hi)
    __bf16* proxyb = (__bf16*)take(nP * 2);
    __bf16* k1     = (__bf16*)take(nP * 2);
    __bf16* keyb   = (__bf16*)take(nP * 2);
    __bf16* v1     = (__bf16*)take(nP * 2);
    __bf16* valT   = (__bf16*)take(nP * 2);

    __bf16* simb  = bufT;                         // [8][HW][256]
    __bf16* simsm = bufT + nSim;                  // softmaxed

    const long long sAct = (long long)C_CH * HW;   // per-batch activation stride
    const long long sPrx = (long long)C_CH * K_REG;

    // 1) fp32 -> bf16 conversions
    cvt_f32_bf16<<<8192, 256, 0, stream>>>(x, bufX, nX);
    cvt_f32_bf16<<<512, 256, 0, stream>>>(W, Wb, nW);
    cvt_f32_bf16<<<512, 256, 0, stream>>>(proxy, proxyb, nP);

    // 2) f_pixel: tmp1 = cba0(xb) ; qT = cba1(tmp1) stored transposed [j][c]
    {
        dim3 g(4, 128, NB);
        gemm_cba<0, 1, __bf16><<<g, 256, 0, stream>>>(
            Wb + 0 * C_CH * C_CH, bufX, bufT, C_CH, HW, C_CH,
            0LL, sAct, sAct, s + 0 * C_CH, b + 0 * C_CH, 1.f);
        gemm_cba<1, 1, __bf16><<<g, 256, 0, stream>>>(
            Wb + 1 * C_CH * C_CH, bufT, bufX, C_CH, HW, C_CH,
            0LL, sAct, sAct, s + 1 * C_CH, b + 1 * C_CH, 1.f);
    }

    // 3) f_object / f_down on proxy (valT stored transposed [k][c])
    {
        dim3 g(4, 2, NB);
        gemm_cba<0, 1, __bf16><<<g, 256, 0, stream>>>(
            Wb + 2 * C_CH * C_CH, proxyb, k1, C_CH, K_REG, C_CH,
            0LL, sPrx, sPrx, s + 2 * C_CH, b + 2 * C_CH, 1.f);
        gemm_cba<0, 1, __bf16><<<g, 256, 0, stream>>>(
            Wb + 3 * C_CH * C_CH, k1, keyb, C_CH, K_REG, C_CH,
            0LL, sPrx, sPrx, s + 3 * C_CH, b + 3 * C_CH, 1.f);
        gemm_cba<0, 1, __bf16><<<g, 256, 0, stream>>>(
            Wb + 4 * C_CH * C_CH, proxyb, v1, C_CH, K_REG, C_CH,
            0LL, sPrx, sPrx, s + 4 * C_CH, b + 4 * C_CH, 1.f);
        gemm_cba<1, 1, __bf16><<<g, 256, 0, stream>>>(
            Wb + 5 * C_CH * C_CH, v1, valT, C_CH, K_REG, C_CH,
            0LL, sPrx, sPrx, s + 5 * C_CH, b + 5 * C_CH, 1.f);
    }

    // 4) sim = (qT x key) * C^-1/2   [HW x 256]
    {
        dim3 g(128, 2, NB);
        gemm_cba<0, 0, __bf16><<<g, 256, 0, stream>>>(
            bufX /*qT*/, keyb, simb, HW, K_REG, C_CH,
            sAct, sPrx, (long long)HW * K_REG,
            nullptr, nullptr, 0.044194173824159216f);
    }

    // 5) softmax over regions (131072 rows of 256)
    softmax256<<<16384, 256, 0, stream>>>(simb, simsm);

    // 6) ctxT = (simsm x valT) stored transposed [c][j]
    {
        dim3 g(128, 4, NB);
        gemm_cba<1, 0, __bf16><<<g, 256, 0, stream>>>(
            simsm, valT, bufX /*ctxT*/, HW, C_CH, K_REG,
            (long long)HW * K_REG, sPrx, sAct,
            nullptr, nullptr, 1.f);
    }

    // 7) f_up: out = cba6(ctxT), fp32 NCHW
    {
        dim3 g(4, 128, NB);
        gemm_cba<0, 1, float><<<g, 256, 0, stream>>>(
            Wb + 6 * C_CH * C_CH, bufX, outp, C_CH, HW, C_CH,
            0LL, sAct, sAct, s + 6 * C_CH, b + 6 * C_CH, 1.f);
    }
}
